// Prefix_Layer_68977174774102
// MI455X (gfx1250) — matrix operand order repaired and checked by reading, not compiled, not run on hardware
//
#include <hip/hip_runtime.h>
#include <hip/hip_bf16.h>

// ---------------- problem constants ----------------
namespace {
constexpr int Dm    = 128;
constexpr int BCn   = 64;
constexpr int BSn   = 512;
constexpr int KWn   = 6;
constexpr int Bn    = 8;
constexpr int Ln    = BCn * BSn;      // 32768
constexpr int NBLKn = Bn * BCn;       // 512
constexpr float SCALEf = 0.08838834764831845f; // 1/sqrt(128)

// workspace layout (byte offsets)
constexpr size_t WS_PREFIX = 0;                          // 8*32768*128 f32 = 134217728 B
constexpr size_t WS_LOGITS = 134217728;                  // 262144 f32
constexpr size_t WS_EX     = WS_LOGITS + 1048576;        // 262144 f32
constexpr size_t WS_PSUM   = WS_EX     + 1048576;        // 262144 f32
constexpr size_t WS_CSUM   = WS_PSUM   + 1048576;        // 8*64*128 f32 = 262144 B
constexpr size_t WS_BPACK  = WS_CSUM   + 262144;         // 8*24*32*16 bf16 = 196608 B
constexpr size_t WS_PBF16  = WS_BPACK  + 196608;         // 8*32768*128 bf16 = 67108864 B

// conv-attn kernel LDS layout (bytes within dynamic smem)
constexpr int IN_STRIDE  = 136;                // bf16 elements per row (128 + 8 pad)
constexpr int SM_IN_OFF  = 0;                  // 512*136*2 = 139264
constexpr int SM_OUT_OFF = 139264;             // 512*136*2 = 139264
constexpr int SM_S_OFF   = 278528;             // f32[512]
constexpr int SM_Q_OFF   = 280576;             // f32[128]
constexpr int SM_RED_OFF = 281088;             // f32[256]
constexpr int SM_AS_OFF  = 282112;             // f32[128]
constexpr int SM_AO_OFF  = 282624;             // f32[128]
constexpr int SM_TOTAL   = 283136;
}

typedef __attribute__((ext_vector_type(16))) __bf16        v16bf;
typedef __attribute__((ext_vector_type(8)))  float         v8f;
typedef __attribute__((ext_vector_type(4)))  unsigned int  v4u;
typedef __attribute__((ext_vector_type(8)))  int           v8i;
typedef __attribute__((ext_vector_type(4)))  int           v4i;

__device__ __forceinline__ unsigned short f2bf(float f) {
  unsigned u = __builtin_bit_cast(unsigned, f);
  unsigned r = u + 0x7FFFu + ((u >> 16) & 1u);   // round-to-nearest-even
  return (unsigned short)(r >> 16);
}
__device__ __forceinline__ float bf2f(unsigned short h) {
  unsigned u = ((unsigned)h) << 16;
  return __builtin_bit_cast(float, u);
}

union FragBF { int4 q2[2]; v16bf v; };

// ---------------------------------------------------------------------------
// Kernel 1: pack conv weights into WMMA 16x16x32 bf16 B-matrix fragments.
// Inner (K) ordering of the conv GEMM: kk = tap*128 + in_ch  (kk in [0,768)).
// B packing here (32x16, 16-bit, contiguous k as in A): lanes 0-15 hold K=0..15, lanes 16-31 hold K=16..31,
// N = lane&15; element j in the lane's v16bf is K_local = 16*(lane>=16)+j.
// ---------------------------------------------------------------------------
__global__ void pack_convw_kernel(const float* __restrict__ conv_w,
                                  unsigned short* __restrict__ bpack) {
  int id = blockIdx.x * 256 + threadIdx.x;       // 8*24*32*16 = 98304 total
  int ot   = id / 12288;
  int rem  = id - ot * 12288;
  int kt   = rem >> 9;
  int rem2 = rem & 511;
  int lane = rem2 >> 4;
  int j    = rem2 & 15;
  int Klocal = ((lane >> 4) << 4) + j;
  int kk = kt * 32 + Klocal;
  int tap = kk >> 7;
  int ic  = kk & 127;
  int oc  = ot * 16 + (lane & 15);
  float v = conv_w[(oc * 128 + ic) * KWn + tap];
  bpack[id] = f2bf(v);
}

// ---------------------------------------------------------------------------
// Kernel 2: logits = silu(x @ w_lw + b_lw)
// ---------------------------------------------------------------------------
__global__ void logits_kernel(const float* __restrict__ x,
                              const float* __restrict__ w_lw,
                              const float* __restrict__ b_lw,
                              float* __restrict__ logits) {
  int tok = blockIdx.x * 256 + threadIdx.x;      // B*L = 262144 tokens
  if (tok >= Bn * Ln) return;
  const float* row = x + (size_t)tok * Dm;
  float z = b_lw[0];
  #pragma unroll 8
  for (int d = 0; d < Dm; ++d) z += row[d] * w_lw[d];
  float s = z / (1.0f + __expf(-z));             // silu
  logits[tok] = s;
}

// ---------------------------------------------------------------------------
// Kernel 3: per-batch scalar scan. For each position t (within a batch):
//   m_t = cummax(l), ex_t = exp(l_t - m_t), psum_t = cumsum(ex).
// Blocked Hillis-Steele scans of width 256 with sequential carry.
// ---------------------------------------------------------------------------
__global__ void scan_kernel(const float* __restrict__ logits,
                            float* __restrict__ ex,
                            float* __restrict__ psum) {
  __shared__ float sm[256];
  int b = blockIdx.x;
  int tid = threadIdx.x;
  float carry_m = -3.402823466e38f;
  float carry_s = 0.0f;
  for (int base = 0; base < Ln; base += 256) {
    int idx = b * Ln + base + tid;
    float v = logits[idx];
    // inclusive max scan
    sm[tid] = v; __syncthreads();
    for (int off = 1; off < 256; off <<= 1) {
      float t = sm[tid];
      if (tid >= off) t = fmaxf(t, sm[tid - off]);
      __syncthreads();
      sm[tid] = t;
      __syncthreads();
    }
    float chunk_max = sm[255];
    float m_incl = fmaxf(carry_m, sm[tid]);
    float e = __expf(v - m_incl);
    __syncthreads();
    // inclusive sum scan of e
    sm[tid] = e; __syncthreads();
    for (int off = 1; off < 256; off <<= 1) {
      float t = sm[tid];
      if (tid >= off) t += sm[tid - off];
      __syncthreads();
      sm[tid] = t;
      __syncthreads();
    }
    float p_incl = carry_s + sm[tid];
    float chunk_sum = sm[255];
    ex[idx]   = e;
    psum[idx] = p_incl;
    carry_m = fmaxf(carry_m, chunk_max);
    carry_s += chunk_sum;
    __syncthreads();
  }
}

// ---------------------------------------------------------------------------
// Kernel 4: per-chunk (512-token) sums of ex * x  -> csum[chunk][d]
// ---------------------------------------------------------------------------
__global__ void chunk_sum_kernel(const float* __restrict__ x,
                                 const float* __restrict__ ex,
                                 float* __restrict__ csum) {
  int chunk = blockIdx.x;            // 0..511 (flat over b*64 + c)
  int d = threadIdx.x;               // 0..127
  float s = 0.0f;
  int gbase = chunk * BSn;
  for (int t = 0; t < BSn; ++t) {
    int gt = gbase + t;
    if (t + 16 < BSn)
      __builtin_prefetch(&x[(size_t)(gt + 16) * Dm + d], 0, 0);
    s += ex[gt] * x[(size_t)gt * Dm + d];
  }
  csum[chunk * Dm + d] = s;
}

// ---------------------------------------------------------------------------
// Kernel 5: exclusive scan over the 64 chunks of each batch
// ---------------------------------------------------------------------------
__global__ void chunk_scan_kernel(float* __restrict__ csum) {
  int b = blockIdx.x;
  int d = threadIdx.x;
  float run = 0.0f;
  for (int c = 0; c < BCn; ++c) {
    int idx = (b * BCn + c) * Dm + d;
    float t = csum[idx];
    csum[idx] = run;
    run += t;
  }
}

// ---------------------------------------------------------------------------
// Kernel 6: prefix_x = (carry + running cumsum(ex*x)) / psum + x
// Also emits a bf16 copy of prefix_x for the TDM-staged WMMA conv.
// ---------------------------------------------------------------------------
__global__ void prefix_final_kernel(const float* __restrict__ x,
                                    const float* __restrict__ ex,
                                    const float* __restrict__ psum,
                                    const float* __restrict__ csum,
                                    float* __restrict__ prefix,
                                    unsigned short* __restrict__ prefix_bf) {
  int chunk = blockIdx.x;
  int d = threadIdx.x;
  float run = csum[chunk * Dm + d];
  int gbase = chunk * BSn;
  for (int t = 0; t < BSn; ++t) {
    int gt = gbase + t;
    float xv = x[(size_t)gt * Dm + d];
    run += ex[gt] * xv;
    float pv = run / psum[gt] + xv;
    prefix[(size_t)gt * Dm + d] = pv;
    prefix_bf[(size_t)gt * Dm + d] = f2bf(pv);
  }
}

// ---------------------------------------------------------------------------
// Kernel 7: fused per-block conv (bf16 WMMA) + maxpool + dual attention +
// fusion GEMV. One workgroup (256 threads = 8 wave32) per block.
// Input staging: Tensor Data Mover 2-D tile DMA (global bf16 -> padded LDS).
// ---------------------------------------------------------------------------
__global__ void conv_attn_kernel(const float* __restrict__ prefix,
                                 const unsigned short* __restrict__ prefix_bf,
                                 const unsigned short* __restrict__ bpack,
                                 const float* __restrict__ conv_b,
                                 const float* __restrict__ fusion_w,
                                 const float* __restrict__ fusion_b,
                                 float* __restrict__ out) {
  extern __shared__ char smem[];
  unsigned short* INb  = (unsigned short*)(smem + SM_IN_OFF);
  unsigned short* OUTb = (unsigned short*)(smem + SM_OUT_OFF);
  float* S   = (float*)(smem + SM_S_OFF);
  float* Q   = (float*)(smem + SM_Q_OFF);
  float* RED = (float*)(smem + SM_RED_OFF);
  float* AS  = (float*)(smem + SM_AS_OFF);
  float* AO  = (float*)(smem + SM_AO_OFF);
  float* XB  = (float*)(smem + SM_IN_OFF);   // reuses IN region after conv

  const int tid  = threadIdx.x;
  const int blk  = blockIdx.x;               // b*64 + c
  const int b    = blk >> 6;
  const int c    = blk & 63;

  // ---- Phase 0: TDM tile load, global bf16 -> LDS (row-padded 256B+16B) ----
  // Tile: tile_dim0 = 128 elems (2B), tile_dim1 = 512 rows; tensor stride 128.
  // LDS padding: pad_interval = 64 DWORDs (enc 5), pad_amount = 4 DWORDs (enc 3)
  //   -> LDS row stride 272 B == IN_STRIDE*2.
  if (tid < 32) {                            // wave 0 only (TDM ignores EXEC)
    unsigned lds_addr = (unsigned)(uintptr_t)(&smem[SM_IN_OFF]);
    unsigned long long gaddr =
        (unsigned long long)(uintptr_t)(prefix_bf + (size_t)blk * BSn * Dm);
    v4u g0;
    g0.x = 1u;                                           // count=1 (valid D#)
    g0.y = lds_addr;                                     // lds_addr[31:0]
    g0.z = (unsigned)(gaddr & 0xFFFFFFFFull);            // global_addr[31:0]
    g0.w = (unsigned)((gaddr >> 32) & 0x1FFFFFFull)      // global_addr[56:32]
         | 0x80000000u;                                  // type=2 ("image")
    v8i g1;
    g1[0] = (int)((1u << 16)        // data_size = 1 -> 2 bytes
                | (1u << 20)        // pad_enable
                | (5u << 22)        // pad_interval: 64 DWORDs
                | (3u << 25));      // pad_amount: 4 DWORDs (16 B)
    g1[1] = (int)(128u << 16);      // tensor_dim0 = 128 (low 16 in [31:16])
    g1[2] = (int)(512u << 16);      // tensor_dim1 = 512 (low 16 in [31:16])
    g1[3] = (int)(128u << 16);      // tile_dim0 = 128
    g1[4] = 512;                    // tile_dim1 = 512
    g1[5] = 128;                    // tensor_dim0_stride = 128 elems
    g1[6] = 0;
    g1[7] = 0;
    v4i gz  = {0, 0, 0, 0};         // groups 2/3 unused (2-D tensor)
    v8i gz8 = {0, 0, 0, 0, 0, 0, 0, 0};
    __builtin_amdgcn_tensor_load_to_lds(g0, g1, gz, gz, gz8, 0);
    __builtin_amdgcn_s_wait_tensorcnt(0);
  }
  __syncthreads();

  // ---- Phase 1: conv as GEMM via v_wmma_f32_16x16x32_bf16 ----
  {
    const int wave = tid >> 5;
    const int lane = tid & 31;
    const int laneHi = lane >> 4;            // 0 for lanes 0-15, 1 for 16-31
    for (int tileIdx = wave; tileIdx < 256; tileIdx += 8) {
      const int tt = tileIdx >> 3;           // t-tile 0..31
      const int ot = tileIdx & 7;            // o-tile 0..7 (== wave)
      v8f acc = {0.f, 0.f, 0.f, 0.f, 0.f, 0.f, 0.f, 0.f};
      for (int kt = 0; kt < 24; ++kt) {
        const int tap = kt >> 2;
        const int row = tt * 16 + (lane & 15) + tap - 2;   // conv input row
        const int ib  = (kt & 3) * 32 + (laneHi << 4);     // bf16 col base
        FragBF fa;
        if (row >= 0 && row < BSn) {
          const int base = row * IN_STRIDE + ib;
          fa.q2[0] = *(const int4*)(&INb[base]);           // K_local {0..7 | 16..23}
          fa.q2[1] = *(const int4*)(&INb[base + 8]);       // K_local {8..15 | 24..31}
        } else {
          fa.q2[0] = make_int4(0, 0, 0, 0);
          fa.q2[1] = make_int4(0, 0, 0, 0);
        }
        FragBF fb;
        const int4* bp = (const int4*)(bpack + (((ot * 24 + kt) * 32 + lane) << 4));
        fb.q2[0] = bp[0];
        fb.q2[1] = bp[1];
        acc = __builtin_amdgcn_wmma_f32_16x16x32_bf16(
            false, fa.v, false, fb.v, (short)0, acc, false, false);
      }
      const int oc = ot * 16 + (lane & 15);
      const float bias = conv_b[oc];
      #pragma unroll
      for (int r = 0; r < 8; ++r) {
        int t = tt * 16 + r + (laneHi << 3);
        OUTb[t * IN_STRIDE + oc] = f2bf(acc[r] + bias);
      }
    }
  }
  __syncthreads();

  // ---- Phase 2: block_repr = max over t ----
  if (tid < Dm) {
    float m = -3.402823466e38f;
    for (int t = 0; t < BSn; ++t)
      m = fmaxf(m, bf2f(OUTb[t * IN_STRIDE + tid]));
    Q[tid] = m;
  }
  __syncthreads();

  // ---- Phase 3: scores over conv outputs ----
  for (int t = tid; t < BSn; t += 256) {
    float s = 0.0f;
    for (int d = 0; d < Dm; ++d)
      s += Q[d] * bf2f(OUTb[t * IN_STRIDE + d]);
    S[t] = s * SCALEf;
  }
  __syncthreads();

  // ---- Phase 4: softmax over 512 scores ----
  float s0 = S[tid], s1 = S[tid + 256];
  RED[tid] = fmaxf(s0, s1); __syncthreads();
  for (int off = 128; off > 0; off >>= 1) {
    if (tid < off) RED[tid] = fmaxf(RED[tid], RED[tid + off]);
    __syncthreads();
  }
  float smax = RED[0]; __syncthreads();
  float e0 = __expf(s0 - smax), e1 = __expf(s1 - smax);
  S[tid] = e0; S[tid + 256] = e1;
  RED[tid] = e0 + e1; __syncthreads();
  for (int off = 128; off > 0; off >>= 1) {
    if (tid < off) RED[tid] += RED[tid + off];
    __syncthreads();
  }
  float inv_s = 1.0f / RED[0];
  __syncthreads();

  // ---- Phase 5: a_s ----
  if (tid < Dm) {
    float a = 0.0f;
    for (int t = 0; t < BSn; ++t)
      a += S[t] * bf2f(OUTb[t * IN_STRIDE + tid]);
    AS[tid] = a * inv_s;
  }
  __syncthreads();

  // ---- Phase 6: stage overlap window (128 tokens, edge-clamped, f32) ----
  for (int idx = tid; idx < 128 * Dm; idx += 256) {
    int row = idx >> 7, col = idx & 127;
    int pos = c * BSn + (BSn - 64) + row;    // c*512 + 448 + row
    if (pos > Ln - 1) pos = Ln - 1;          // edge pad within batch
    XB[row * Dm + col] = prefix[((size_t)b * Ln + pos) * Dm + col];
  }
  __syncthreads();

  // ---- Phase 7: overlap attention ----
  if (tid < 128) {
    float s = 0.0f;
    for (int d = 0; d < Dm; ++d) s += Q[d] * XB[tid * Dm + d];
    S[tid] = s * SCALEf;
  }
  __syncthreads();
  if (tid < 128) RED[tid] = S[tid];
  __syncthreads();
  for (int off = 64; off > 0; off >>= 1) {
    if (tid < off) RED[tid] = fmaxf(RED[tid], RED[tid + off]);
    __syncthreads();
  }
  float m2 = RED[0]; __syncthreads();
  if (tid < 128) {
    float e = __expf(S[tid] - m2);
    S[tid] = e;
    RED[tid] = e;
  }
  __syncthreads();
  for (int off = 64; off > 0; off >>= 1) {
    if (tid < off) RED[tid] += RED[tid + off];
    __syncthreads();
  }
  float inv2 = 1.0f / RED[0];
  __syncthreads();
  if (tid < Dm) {
    float a = 0.0f;
    for (int t = 0; t < 128; ++t) a += S[t] * XB[t * Dm + tid];
    AO[tid] = a * inv2;
  }
  __syncthreads();

  // ---- Phase 8: fusion GEMV: [a_s, a_o, block_repr] @ fusion_w + b ----
  if (tid < Dm) {
    float acc = fusion_b[tid];
    for (int d = 0; d < Dm; ++d) acc += AS[d] * fusion_w[d * Dm + tid];
    for (int d = 0; d < Dm; ++d) acc += AO[d] * fusion_w[(Dm + d) * Dm + tid];
    for (int d = 0; d < Dm; ++d) acc += Q[d]  * fusion_w[(2 * Dm + d) * Dm + tid];
    out[(size_t)blk * Dm + tid] = acc;
    out[(size_t)NBLKn * Dm + (size_t)blk * Dm + tid] = acc;  // tuple duplicate
  }
}

// ---------------------------------------------------------------------------
extern "C" void kernel_launch(void* const* d_in, const int* in_sizes, int n_in,
                              void* d_out, int out_size, void* d_ws, size_t ws_size,
                              hipStream_t stream) {
  (void)in_sizes; (void)n_in; (void)out_size; (void)ws_size;
  const float* x    = (const float*)d_in[0];
  const float* w_lw = (const float*)d_in[1];
  const float* b_lw = (const float*)d_in[2];
  const float* cw   = (const float*)d_in[3];
  const float* cb   = (const float*)d_in[4];
  const float* fw   = (const float*)d_in[5];
  const float* fb   = (const float*)d_in[6];
  float* out = (float*)d_out;

  char* ws = (char*)d_ws;
  float*          prefix    = (float*)(ws + WS_PREFIX);
  float*          logits    = (float*)(ws + WS_LOGITS);
  float*          ex        = (float*)(ws + WS_EX);
  float*          psum      = (float*)(ws + WS_PSUM);
  float*          csum      = (float*)(ws + WS_CSUM);
  unsigned short* bpack     = (unsigned short*)(ws + WS_BPACK);
  unsigned short* prefix_bf = (unsigned short*)(ws + WS_PBF16);

  pack_convw_kernel<<<dim3(384), dim3(256), 0, stream>>>(cw, bpack);
  logits_kernel<<<dim3((Bn * Ln) / 256), dim3(256), 0, stream>>>(x, w_lw, b_lw, logits);
  scan_kernel<<<dim3(Bn), dim3(256), 0, stream>>>(logits, ex, psum);
  chunk_sum_kernel<<<dim3(NBLKn), dim3(Dm), 0, stream>>>(x, ex, csum);
  chunk_scan_kernel<<<dim3(Bn), dim3(Dm), 0, stream>>>(csum);
  prefix_final_kernel<<<dim3(NBLKn), dim3(Dm), 0, stream>>>(x, ex, psum, csum,
                                                            prefix, prefix_bf);
  conv_attn_kernel<<<dim3(NBLKn), dim3(256), SM_TOTAL, stream>>>(
      prefix, prefix_bf, bpack, cb, fw, fb, out);
}
